// DecoderSequence_27943057228140
// MI455X (gfx1250) — matrix operand order repaired and checked by reading, not compiled, not run on hardware
//
#include <hip/hip_runtime.h>
#include <hip/hip_bf16.h>
#include <math.h>

// Problem constants (match reference)
#define BB 16
#define TT 2048
#define DD 1024
#define H3 3072   // 3*DD

typedef __attribute__((ext_vector_type(16))) __bf16 v16bf;
typedef __attribute__((ext_vector_type(8)))  float  v8f;
typedef unsigned int u32x4 __attribute__((ext_vector_type(4)));
typedef int          i32x4 __attribute__((ext_vector_type(4)));
typedef int          i32x8 __attribute__((ext_vector_type(8)));
typedef float        f32x4 __attribute__((ext_vector_type(4)));

union Frag16 {
  v16bf v;
  unsigned int u[8];
  u32x4 q[2];          // two 16B halves -> ds_load_b128 pairs
};

// ---- TDM availability / arity detection -----------------------------------
#if __has_include(<hip/amd_detail/amd_gfx1250_TDM.h>)
#define TDM_6ARG 1
#else
#define TDM_6ARG 0
#endif

#if __has_builtin(__builtin_amdgcn_tensor_load_to_lds) && \
    __has_builtin(__builtin_amdgcn_s_wait_tensorcnt)
#define HAVE_TDM 1
#else
#define HAVE_TDM 0
#endif

#if HAVE_TDM
// Tensor DMA: global -> LDS, up to 3D. Units: 4-byte dwords (data_size=2).
__device__ __forceinline__ void tdm_load_to_lds(unsigned lds_off, const void* gptr,
                                                unsigned t0_dw, unsigned rows, unsigned planes,
                                                unsigned long long stride0_dw,
                                                unsigned long long stride1_dw) {
  unsigned long long ga = (unsigned long long)gptr;
  u32x4 g0;
  g0[0] = 1u;                                                  // count=1, user D#
  g0[1] = lds_off;                                             // lds_addr (bytes)
  g0[2] = (unsigned)ga;                                        // global_addr[31:0]
  g0[3] = ((unsigned)(ga >> 32) & 0x01FFFFFFu) | (2u << 30);   // addr[56:32] | type=2

  const unsigned td0 = t0_dw;           // tensor_dim0 == tile width (no OOB)
  const unsigned td1 = rows;            // tensor_dim1
  i32x8 g1;
  g1[0] = (int)(2u << 16);                                        // data_size=4B
  g1[1] = (int)((td0 & 0xFFFFu) << 16);                           // tensor_dim0 lo16
  g1[2] = (int)((td0 >> 16) | ((td1 & 0xFFFFu) << 16));           // dim0 hi | dim1 lo
  g1[3] = (int)((td1 >> 16) | ((t0_dw & 0xFFFFu) << 16));         // dim1 hi | tile_dim0
  g1[4] = (int)(((rows > 1 ? rows : 0u) & 0xFFFFu) |              // tile_dim1
                (((planes > 1 ? planes : 0u) & 0xFFFFu) << 16));  // tile_dim2
  g1[5] = (int)(unsigned)(stride0_dw & 0xFFFFFFFFull);            // dim0_stride lo32
  g1[6] = (int)((unsigned)((stride0_dw >> 32) & 0xFFFFull) |      // dim0_stride hi16
                ((unsigned)(stride1_dw & 0xFFFFull) << 16));      // dim1_stride lo16
  g1[7] = (int)(unsigned)((stride1_dw >> 16) & 0xFFFFFFFFull);    // dim1_stride [47:16]

  i32x4 g2 = {};
  g2[0] = (int)(planes ? planes : 1u);                            // tensor_dim2
  i32x4 g3 = {};
#if TDM_6ARG
  i32x8 gx = {};
  __builtin_amdgcn_tensor_load_to_lds(g0, g1, g2, g3, gx, 0);
#else
  __builtin_amdgcn_tensor_load_to_lds(g0, g1, g2, g3, 0);
#endif
}
#endif  // HAVE_TDM

__device__ __forceinline__ unsigned short f2bf(float f) {
  unsigned int u = __float_as_uint(f);
  unsigned int r = u + 0x7FFFu + ((u >> 16) & 1u);   // round-to-nearest-even
  return (unsigned short)(r >> 16);
}

__device__ __forceinline__ float sigmoidf_(float x) {
  return 1.0f / (1.0f + __expf(-x));
}

extern __shared__ char dsmem[];

// ---------------------------------------------------------------------------
// prep: convert weights fp32 -> bf16, zero h state, reset grid barrier
// ---------------------------------------------------------------------------
__global__ void prep_kernel(const float* __restrict__ w_ih, const float* __restrict__ w_hh,
                            unsigned short* __restrict__ wih_bf, unsigned short* __restrict__ whh_bf,
                            unsigned short* __restrict__ h_bf, float* __restrict__ h_f32,
                            unsigned int* __restrict__ bar) {
  size_t i = (size_t)blockIdx.x * blockDim.x + threadIdx.x;
  size_t stride = (size_t)gridDim.x * blockDim.x;
  const size_t NW = (size_t)H3 * DD;
  for (size_t k = i; k < NW; k += stride) {
    wih_bf[k] = f2bf(w_ih[k]);
    whh_bf[k] = f2bf(w_hh[k]);
  }
  for (size_t k = i; k < 2u * BB * DD; k += stride) h_bf[k] = 0;
  for (size_t k = i; k < (size_t)BB * DD; k += stride) h_f32[k] = 0.0f;
  if (i < 8) bar[i] = 0u;
}

// ---------------------------------------------------------------------------
// xproj: xg[M=32768, N=3072] = bf16(x) @ bf16(w_ih)^T + b_ih   (fp32 out)
// 128 threads = 4 waves, workgroup tile 64x64, wave tile 32x32 (2x2 WMMA), K-step 32.
// Dynamic LDS: [0,4096) A tile bf16, [4096,8192) B tile bf16 (TDM-staged).
// ---------------------------------------------------------------------------
__global__ void __launch_bounds__(128)
xproj_kernel(const float* __restrict__ x, const unsigned short* __restrict__ wih_bf,
             const float* __restrict__ b_ih, float* __restrict__ xg) {
  unsigned short* lds_a = (unsigned short*)dsmem;            // [64][32] bf16
  unsigned short* lds_b = (unsigned short*)(dsmem + 4096);   // [64][32] bf16

  const int tid  = threadIdx.x;
  const int lane = tid & 31;
  const int wave = tid >> 5;              // 0..3
  const int wm   = (wave >> 1) * 32;      // wave M offset in tile
  const int wn   = (wave & 1) * 32;       // wave N offset in tile
  const int m0   = blockIdx.x * 64;       // global row base
  const int n0   = blockIdx.y * 64;       // global col base
  const int ln   = lane & 15;
  const int lh   = lane >> 4;

  v8f acc[2][2];
  acc[0][0] = (v8f){}; acc[0][1] = (v8f){}; acc[1][0] = (v8f){}; acc[1][1] = (v8f){};

  for (int k0 = 0; k0 < DD; k0 += 32) {
    __syncthreads();
#if HAVE_TDM
    // B tile via TDM: 16 dwords (32 bf16) x 64 rows, row stride 512 dwords
    if (wave == 0) {
      tdm_load_to_lds(4096u, wih_bf + (size_t)n0 * DD + k0, 16u, 64u, 1u, 512ull, 0ull);
    }
#endif
    // stage A (convert fp32 -> bf16) -- overlaps the TDM transfer
    for (int idx = tid; idx < 64 * 32; idx += 128) {
      int i = idx >> 5, k = idx & 31;
      lds_a[idx] = f2bf(x[(size_t)(m0 + i) * DD + k0 + k]);
    }
#if HAVE_TDM
    if (wave == 0) __builtin_amdgcn_s_wait_tensorcnt(0);
#else
    {
      const unsigned int* wsrc = (const unsigned int*)wih_bf;
      unsigned int* dst = (unsigned int*)lds_b;
      for (int idx = tid; idx < 64 * 16; idx += 128) {
        int i = idx >> 4, kd = idx & 15;
        dst[i * 16 + kd] = wsrc[(size_t)(n0 + i) * (DD / 2) + (k0 >> 1) + kd];
      }
    }
#endif
    __syncthreads();

    const unsigned int* a32 = (const unsigned int*)lds_a;
    const unsigned int* b32 = (const unsigned int*)lds_b;
    #pragma unroll
    for (int mt = 0; mt < 2; ++mt) {
      // A 16x32 bf16 fragment (ISA layout): per lane two 16B chunks at +0,+32B
      Frag16 a;
      const int arow = wm + mt * 16 + ln;
      const u32x4* aq = (const u32x4*)(a32 + arow * 16 + lh * 4);
      a.q[0] = aq[0];
      a.q[1] = aq[2];
      #pragma unroll
      for (int nt = 0; nt < 2; ++nt) {
        // B 32x16 bf16 fragment (ISA layout): per lane two 16B chunks at +0,+32B
        Frag16 b;
        const int bcol = wn + nt * 16 + ln;
        const u32x4* bq = (const u32x4*)(b32 + bcol * 16 + lh * 4);
        b.q[0] = bq[0];
        b.q[1] = bq[2];
        acc[mt][nt] = __builtin_amdgcn_wmma_f32_16x16x32_bf16(
            false, a.v, false, b.v, (short)0, acc[mt][nt], false, false);
      }
    }
  }

  // epilogue: + b_ih, store fp32
  #pragma unroll
  for (int mt = 0; mt < 2; ++mt) {
    #pragma unroll
    for (int nt = 0; nt < 2; ++nt) {
      #pragma unroll
      for (int v = 0; v < 8; ++v) {
        int m = m0 + wm + mt * 16 + v + 8 * lh;  // C layout: M = vgpr + 8*(lane>=16)
        int n = n0 + wn + nt * 16 + ln;          //           N = lane%16
        xg[(size_t)m * H3 + n] = acc[mt][nt][v] + b_ih[n];
      }
    }
  }
}

// ---------------------------------------------------------------------------
// GRU scan: persistent grid, 96 blocks x 192 threads (6 waves).
// Dynamic LDS layout:
//   [0,       196608)  W_hh slice  [3 gates][32 cols][1024 k] bf16 (192KB, TDM, resident)
//   [196608,  229376)  h stage     [16][1024] bf16 (32KB, TDM per step)
//   [229376,  235520)  hg exchange [3][16][32] f32
// ---------------------------------------------------------------------------
#define LDS_W_OFF  0u
#define LDS_H_OFF  196608u
#define LDS_HG_OFF 229376u

__global__ void __launch_bounds__(192)
gru_scan_kernel(const float* __restrict__ xg, const unsigned short* __restrict__ whh_bf,
                const float* __restrict__ b_hh, unsigned short* __restrict__ h_bf,
                float* __restrict__ h_f32, float* __restrict__ ys,
                unsigned int* __restrict__ bar) {
  unsigned short* lds_w  = (unsigned short*)(dsmem + LDS_W_OFF);
  unsigned short* lds_h  = (unsigned short*)(dsmem + LDS_H_OFF);
  float*          lds_hg = (float*)(dsmem + LDS_HG_OFF);

  const int tid  = threadIdx.x;
  const int lane = tid & 31;
  const int wave = tid >> 5;          // 0..5
  const int g    = wave >> 1;         // gate 0..2
  const int nt   = wave & 1;          // n-tile 0..1
  const int ln   = lane & 15;
  const int lh   = lane >> 4;
  const int c0   = blockIdx.x * 32;   // this block's column base within H

  // one-time: W_hh slice -> LDS.
#if HAVE_TDM
  // 3D TDM: 512 dw/row x 32 rows x 3 gate planes; row stride 512 dw,
  // plane stride DD*DD bf16 = 524288 dw. Lands gate-major contiguous in LDS.
  if (wave == 0) {
    tdm_load_to_lds(LDS_W_OFF, whh_bf + (size_t)c0 * DD,
                    512u, 32u, 3u, 512ull, 524288ull);
    __builtin_amdgcn_s_wait_tensorcnt(0);
  }
#else
  {
    const unsigned int* wsrc = (const unsigned int*)whh_bf;
    unsigned int* dst = (unsigned int*)lds_w;
    for (int idx = tid; idx < 96 * 512; idx += 192) {
      int rr = idx >> 9, kd = idx & 511;
      int gg = rr >> 5, cc = rr & 31;
      dst[idx] = wsrc[(size_t)(gg * DD + c0 + cc) * (DD / 2) + kd];
    }
  }
#endif

  int p = 0;
  for (int t = 0; t < TT; ++t) {
    __syncthreads();
    // stage h_bf[p] -> LDS (32KB)
#if HAVE_TDM
    if (wave == 0) {
      tdm_load_to_lds(LDS_H_OFF, h_bf + (size_t)p * BB * DD, 512u, 16u, 1u, 512ull, 0ull);
      __builtin_amdgcn_s_wait_tensorcnt(0);
    }
#else
    {
      const unsigned int* hsrc = (const unsigned int*)(h_bf + (size_t)p * BB * DD);
      unsigned int* dst = (unsigned int*)lds_h;
      for (int idx = tid; idx < BB * 512; idx += 192) dst[idx] = hsrc[idx];
    }
#endif
    __syncthreads();

    // hg tile for (gate g, n-tile nt): 16x16 over K=1024
    v8f acc = (v8f){};
    const unsigned int* a32 = (const unsigned int*)lds_h;
    const unsigned int* brow = ((const unsigned int*)lds_w) + (size_t)(g * 32 + nt * 16 + ln) * 512;
    for (int ks = 0; ks < 32; ++ks) {
      Frag16 a, b;
      const u32x4* aq = (const u32x4*)(a32 + ln * 512 + ks * 16 + lh * 4);
      a.q[0] = aq[0];
      a.q[1] = aq[2];            // +32B
      const u32x4* bq = (const u32x4*)(brow + ks * 16 + lh * 4);
      b.q[0] = bq[0];
      b.q[1] = bq[2];
      acc = __builtin_amdgcn_wmma_f32_16x16x32_bf16(
          false, a.v, false, b.v, (short)0, acc, false, false);
    }

    // park hg (+ b_hh) in LDS for cross-wave gate combination
    #pragma unroll
    for (int v = 0; v < 8; ++v) {
      int m = v + 8 * lh;
      int n = nt * 16 + ln;
      lds_hg[(g * 16 + m) * 32 + n] = acc[v] + b_hh[g * DD + c0 + n];
    }
    __syncthreads();

    // elementwise gate math for this block's 16x32 slice
    for (int e = tid; e < BB * 32; e += 192) {
      int m = e >> 5, c = e & 31;
      size_t xrow = (size_t)(m * TT + t) * H3;
      float xr = xg[xrow + 0 * DD + c0 + c];
      float xz = xg[xrow + 1 * DD + c0 + c];
      float xn = xg[xrow + 2 * DD + c0 + c];
      // prefetch next timestep's xg cache lines (128B covers the 32-col slice)
      if (c == 0 && t + 1 < TT) {
        size_t xrow_n = (size_t)(m * TT + t + 1) * H3;
        __builtin_prefetch(&xg[xrow_n + 0 * DD + c0], 0, 0);
        __builtin_prefetch(&xg[xrow_n + 1 * DD + c0], 0, 0);
        __builtin_prefetch(&xg[xrow_n + 2 * DD + c0], 0, 0);
      }
      float hr = lds_hg[(0 * 16 + m) * 32 + c];
      float hz = lds_hg[(1 * 16 + m) * 32 + c];
      float hn = lds_hg[(2 * 16 + m) * 32 + c];
      float r = sigmoidf_(xr + hr);
      float z = sigmoidf_(xz + hz);
      float n = tanhf(xn + r * hn);
      float hprev = h_f32[(size_t)m * DD + c0 + c];
      float hnew = (1.0f - z) * n + z * hprev;
      h_f32[(size_t)m * DD + c0 + c] = hnew;
      h_bf[(size_t)(p ^ 1) * BB * DD + (size_t)m * DD + c0 + c] = f2bf(hnew);
      ys[(size_t)(m * TT + t) * DD + c0 + c] = hnew;
    }

    // device-wide generation barrier (bar[0]=count, bar[1]=generation)
    __threadfence();
    __syncthreads();
    if (tid == 0) {
      unsigned arrived = atomicAdd(&bar[0], 1u);
      if (arrived == gridDim.x - 1) {
        bar[0] = 0u;
        __threadfence();
        atomicAdd(&bar[1], 1u);
      } else {
        while (__hip_atomic_load(&bar[1], __ATOMIC_ACQUIRE, __HIP_MEMORY_SCOPE_AGENT)
               <= (unsigned)t) {
          __builtin_amdgcn_s_sleep(2);
        }
      }
    }
    __syncthreads();
    p ^= 1;
  }
}

// ---------------------------------------------------------------------------
// gather valid timesteps + L2 normalize rows into d_out (float4 vectorized)
// one block per (b,t); prefix over 16 lengths computed in-kernel
// ---------------------------------------------------------------------------
__global__ void __launch_bounds__(256)
gather_norm_kernel(const float* __restrict__ ys, const int* __restrict__ lengths,
                   float* __restrict__ out) {
  int blk = blockIdx.x;
  int b = blk >> 11;          // / 2048
  int t = blk & (TT - 1);
  int len = lengths[b];
  if (t >= len) return;

  long off = 0;
  for (int i = 0; i < b; ++i) off += lengths[i];

  const f32x4* row = (const f32x4*)(ys + (size_t)(b * TT + t) * DD);  // 256 vec4
  f32x4 v = row[threadIdx.x];
  float ss = v[0] * v[0] + v[1] * v[1] + v[2] * v[2] + v[3] * v[3];

  __shared__ float red[256];
  red[threadIdx.x] = ss;
  __syncthreads();
  for (int s = 128; s > 0; s >>= 1) {
    if (threadIdx.x < s) red[threadIdx.x] += red[threadIdx.x + s];
    __syncthreads();
  }
  float scale = 1.0f / fmaxf(sqrtf(red[0]), 1e-12f);
  f32x4* orow = (f32x4*)(out + (size_t)(off + t) * DD);
  f32x4 o;
  o[0] = v[0] * scale; o[1] = v[1] * scale; o[2] = v[2] * scale; o[3] = v[3] * scale;
  orow[threadIdx.x] = o;
}

// ---------------------------------------------------------------------------
extern "C" void kernel_launch(void* const* d_in, const int* in_sizes, int n_in,
                              void* d_out, int out_size, void* d_ws, size_t ws_size,
                              hipStream_t stream) {
  (void)in_sizes; (void)n_in; (void)out_size; (void)ws_size;
  const float* x       = (const float*)d_in[0];
  const int*   lengths = (const int*)d_in[1];
  const float* w_ih    = (const float*)d_in[2];
  const float* w_hh    = (const float*)d_in[3];
  const float* b_ih    = (const float*)d_in[4];
  const float* b_hh    = (const float*)d_in[5];
  float* out = (float*)d_out;

  // workspace carve-up (256B aligned)
  size_t off = 0;
  auto carve = [&](size_t bytes) -> void* {
    void* p = (char*)d_ws + off;
    off += (bytes + 255) & ~(size_t)255;
    return p;
  };
  float*          xg     = (float*)carve((size_t)BB * TT * H3 * 4);        // 402 MB
  float*          ys     = (float*)carve((size_t)BB * TT * DD * 4);        // 128 MB
  unsigned short* wih_bf = (unsigned short*)carve((size_t)H3 * DD * 2);    // 6 MB
  unsigned short* whh_bf = (unsigned short*)carve((size_t)H3 * DD * 2);    // 6 MB
  unsigned short* h_bf   = (unsigned short*)carve((size_t)2 * BB * DD * 2);
  float*          h_f32  = (float*)carve((size_t)BB * DD * 4);
  unsigned int*   bar    = (unsigned int*)carve(256);

  prep_kernel<<<1024, 256, 0, stream>>>(w_ih, w_hh, wih_bf, whh_bf, h_bf, h_f32, bar);

  dim3 ggrid((BB * TT) / 64, H3 / 64);   // 512 x 48
  xproj_kernel<<<ggrid, 128, 8192, stream>>>(x, wih_bf, b_ih, xg);

  const size_t scan_lds = (size_t)LDS_HG_OFF + (size_t)3 * 16 * 32 * 4;    // 235,520 B
  gru_scan_kernel<<<H3 / 32, 192, scan_lds, stream>>>(xg, whh_bf, b_hh, h_bf, h_f32, ys, bar);

  gather_norm_kernel<<<BB * TT, 256, 0, stream>>>(ys, lengths, out);
}
